// MultiModalVAE_46059229283218
// MI455X (gfx1250) — matrix operand order repaired and checked by reading, not compiled, not run on hardware
//
#include <hip/hip_runtime.h>
#include <math.h>

// ---------------------------------------------------------------------------
// Types for CDNA5 WMMA (wave32): 16x16x32 bf16 -> f32
// ---------------------------------------------------------------------------
typedef __bf16 bf16_t;
typedef bf16_t v16bf __attribute__((ext_vector_type(16)));
typedef float  v8f   __attribute__((ext_vector_type(8)));

#define WPB 8      // waves per block (wave32)
#define NTHREADS 256

// Dimensions (fixed by the reference)
#define BB 16
#define MM 4
#define SS 512
#define DD 1024
#define HH 8
#define HD 128

// Async global->LDS staging (CDNA5 GLOBAL_LOAD_ASYNC_TO_LDS_B128, ASYNCcnt)
#if __has_builtin(__builtin_amdgcn_global_load_async_to_lds_b128)
#define CDNA5_ASYNC_LDS 1
#else
#define CDNA5_ASYNC_LDS 0
#endif

struct GemmMap { int rows_per_b; int mod; int mapped; };

__device__ __forceinline__ long map_row(const GemmMap m, int r, int ld) {
  if (!m.mapped) return (long)r * (long)ld;
  int b = r / m.rows_per_b;
  int s = r - b * m.rows_per_b;
  return ((long)((b * MM + m.mod) * SS + s)) * (long)ld;
}

__device__ __forceinline__ bf16_t cvt_bf(float x)  { return (bf16_t)x; }
__device__ __forceinline__ bf16_t cvt_bf(bf16_t x) { return x; }

__device__ __forceinline__ float sigmoidf_(float x) { return 1.0f / (1.0f + expf(-x)); }

__device__ __forceinline__ float gelu_tanh(float x) {
  float x3 = x * x * x;
  return 0.5f * x * (1.0f + tanhf(0.7978845608028654f * (x + 0.044715f * x3)));
}

__device__ __forceinline__ float wave_sum(float v) {
#pragma unroll
  for (int m = 16; m > 0; m >>= 1) v += __shfl_xor(v, m, 32);
  return v;
}

__device__ __forceinline__ void wait_async_then_barrier() {
#if CDNA5_ASYNC_LDS
#if __has_builtin(__builtin_amdgcn_s_wait_asynccnt)
  __builtin_amdgcn_s_wait_asynccnt(0);
#else
  asm volatile("s_wait_asynccnt 0x0" ::: "memory");
#endif
#endif
  __syncthreads();
}

// A fragment: 16x32 (MxK) bf16. Lane L<16 holds row M=L, K in {kb+0..15};
// lane L>=16 holds row M=L-16, K in {kb+16..31} (matches packed B order).
template<typename AT>
__device__ __forceinline__ v16bf load_a(const AT* __restrict__ A, long roff, int kb, int half) {
  v16bf f;
  const AT* p = A + roff + kb + half * 16;
#pragma unroll
  for (int j = 0; j < 8; ++j) {
    f[j]     = cvt_bf(p[j]);
    f[j + 8] = cvt_bf(p[j + 8]);
  }
  return f;
}

// ---------------------------------------------------------------------------
// Packed-weight layout: W[K,N] f32 -> bf16 tiles of 32(K) x 64(N).
// Tile order: n-strip major, k-tile minor (k streams contiguously):
//   tile_base = (nt * (K/32) + kt) * 2048
// Within a tile, elements are stored in WMMA B-fragment order:
//   offset = f*512 + lane*16 + e   with  f = (c&63)>>4, lane = (k&16) + (c&15),
//   e = k&15   (lane holds col c, elements e = K within its half; contiguous k order, as in load_a)
// ---------------------------------------------------------------------------
__global__ __launch_bounds__(NTHREADS)
void pack_w_kernel(const float* __restrict__ W, bf16_t* __restrict__ P, int K, int N) {
  int d = blockIdx.x * NTHREADS + threadIdx.x;   // packed destination index
  int total = K * N;
  if (d >= total) return;
  int tile = d >> 11;            // / 2048
  int rem  = d & 2047;
  int f    = rem >> 9;           // 0..3
  int lane = (rem >> 4) & 31;    // 0..31
  int e    = rem & 15;           // 0..15
  int ktiles = K >> 5;
  int nt   = tile / ktiles;
  int kt   = tile - nt * ktiles;
  int k = (kt << 5) + ((lane >> 4) << 4) + e;
  int c = (nt << 6) + (f << 4) + (lane & 15);
  P[d] = (bf16_t)W[(size_t)k * N + c];
}

// Cooperative staging of one packed 4 KB W tile into LDS (linear copy).
__device__ __forceinline__ void stage_w(const bf16_t* __restrict__ Wpk, int ktiles,
                                        int kt, int nt, bf16_t* __restrict__ dstTile,
                                        int tid) {
  const bf16_t* src = Wpk + (((size_t)nt * ktiles + kt) << 11) + tid * 8;
  bf16_t* dst = dstTile + tid * 8;
#if CDNA5_ASYNC_LDS
  typedef int gv4i __attribute__((vector_size(16)));
  typedef __attribute__((address_space(1))) gv4i* gv4i_g;
  typedef __attribute__((address_space(3))) gv4i* gv4i_s;
  __builtin_amdgcn_global_load_async_to_lds_b128((gv4i_g)(src), (gv4i_s)(dst), 0, 0);
#else
  *(uint4*)dst = *(const uint4*)src;
#endif
}

// ---------------------------------------------------------------------------
// WMMA GEMM: C[R,N] = A[R,K] @ W[K,N] (+ bias), W packed bf16.
//   MODE 0: f32 out + bias
//   MODE 1: f32 out + bias + tanh-GELU
//   MODE 2: bf16 out + bias
//   MODE 3: f32 out + bias, zero rows with s >= seqlen (s = row % rows_per_b)
// Block = 8 waves = 256 rows x 64 cols; wave = 32x64 (2 A-frags x 4 B-frags).
// Packed W tile double-buffered in LDS via async-to-LDS; B-frags are single
// aligned 32-byte LDS reads (2x ds_load_b128 each).
// ---------------------------------------------------------------------------
template<typename AT, int MODE>
__global__ __launch_bounds__(NTHREADS)
void gemm_wmma(const AT* __restrict__ A, int lda, GemmMap amap,
               const bf16_t* __restrict__ Wpk, const float* __restrict__ bias,
               void* __restrict__ Cptr, int ldc, GemmMap cmap,
               int R, int N, int K, int seqlen)
{
  __shared__ __align__(32) bf16_t wtile[2][2048];

  const int tid  = threadIdx.x;
  const int lane = tid & 31;
  const int wave = tid >> 5;
  const int ntiles = N >> 6;
  const int ktiles = K >> 5;
  const int mb = blockIdx.x / ntiles;       // 256-row macro block
  const int nt = blockIdx.x - mb * ntiles;  // shared by all waves in block
  const int half = lane >> 4;
  const int rlow = lane & 15;
  const int nbase = nt << 6;

  v8f acc[2][4];
#pragma unroll
  for (int i = 0; i < 2; ++i)
#pragma unroll
    for (int f = 0; f < 4; ++f) acc[i][f] = {};

  int r0 = mb * 256 + wave * 32 + rlow;
  int r1 = r0 + 16;
  int r0c = (r0 < R) ? r0 : (R - 1);
  int r1c = (r1 < R) ? r1 : (R - 1);
  long a0off = map_row(amap, r0c, lda);
  long a1off = map_row(amap, r1c, lda);

  // Prologue: stage first packed W tile
  stage_w(Wpk, ktiles, 0, nt, wtile[0], tid);
  wait_async_then_barrier();

  int buf = 0;
  for (int kt = 0; kt < ktiles; ++kt) {
    if (kt + 1 < ktiles) stage_w(Wpk, ktiles, kt + 1, nt, wtile[buf ^ 1], tid);

    const int kb = kt << 5;
    v16bf a0 = load_a<AT>(A, a0off, kb, half);
    v16bf a1 = load_a<AT>(A, a1off, kb, half);
#pragma unroll
    for (int f = 0; f < 4; ++f) {
      v16bf bfrag = *(const v16bf*)(wtile[buf] + (f << 9) + (lane << 4));
      acc[0][f] = __builtin_amdgcn_wmma_f32_16x16x32_bf16(
          false, a0, false, bfrag, (short)0, acc[0][f], false, false);
      acc[1][f] = __builtin_amdgcn_wmma_f32_16x16x32_bf16(
          false, a1, false, bfrag, (short)0, acc[1][f], false, false);
    }
    wait_async_then_barrier();
    buf ^= 1;
  }

  // Epilogue. C/D layout: lane holds col = rlow within frag; VGPR rr -> row
  // rr + (lane>=16 ? 8 : 0) within the 16-row frag.  (ISA 7.12.2)
#pragma unroll
  for (int i = 0; i < 2; ++i) {
#pragma unroll
    for (int f = 0; f < 4; ++f) {
      int c = nbase + f * 16 + rlow;
      float bvv = bias ? bias[c] : 0.0f;
#pragma unroll
      for (int rr = 0; rr < 8; ++rr) {
        int row = mb * 256 + wave * 32 + i * 16 + half * 8 + rr;
        if (row < R) {
          float v = acc[i][f][rr] + bvv;
          long co = map_row(cmap, row, ldc) + c;
          if (MODE == 0) {
            ((float*)Cptr)[co] = v;
          } else if (MODE == 1) {
            ((float*)Cptr)[co] = gelu_tanh(v);
          } else if (MODE == 2) {
            ((bf16_t*)Cptr)[co] = (bf16_t)v;
          } else {
            int s = row % cmap.rows_per_b;
            ((float*)Cptr)[co] = (s < seqlen) ? v : 0.0f;
          }
        }
      }
    }
  }
}

// ---------------------------------------------------------------------------
// Gate scalar + modulation: one wave per valid row (b, s) of modality `mod`.
// ---------------------------------------------------------------------------
__global__ __launch_bounds__(NTHREADS)
void gate_mod_kernel(const float* __restrict__ P, const float* __restrict__ G1,
                     const float* __restrict__ gw2, const float* __restrict__ gb2,
                     const float* __restrict__ alphas, const float* __restrict__ me,
                     bf16_t* __restrict__ ST, int s_i, int mod, int R)
{
  const int lane = threadIdx.x & 31;
  const int wave = threadIdx.x >> 5;
  int idx = blockIdx.x * WPB + wave;
  if (idx >= R) return;
  int b = idx / s_i;
  int s = idx - b * s_i;
  long row = (long)((b * MM + mod) * SS + s);

  const float* g1 = G1 + row * 256;
  const float* gw = gw2 + mod * 256;
  float part = 0.0f;
#pragma unroll
  for (int j = 0; j < 8; ++j) part += g1[lane * 8 + j] * gw[lane * 8 + j];
  part = wave_sum(part);

  float g = sigmoidf_(part + gb2[mod]);
  float a = sigmoidf_(alphas[mod]);
  float scale = g * a + (1.0f - a);

  const float* prow = P + row * DD;
  bf16_t* strow = ST + row * DD;
  const float* merow = me + mod * DD;
  for (int c = lane; c < DD; c += 32)
    strow[c] = (bf16_t)(prow[c] * scale + merow[c]);
}

// ---------------------------------------------------------------------------
// Per-position cross-modality attention: one wave per (b, h, s).
// ---------------------------------------------------------------------------
__global__ __launch_bounds__(NTHREADS)
void attn_kernel(const bf16_t* __restrict__ Q, const bf16_t* __restrict__ K,
                 const bf16_t* __restrict__ V, bf16_t* __restrict__ AO,
                 const float* __restrict__ betas, const float* __restrict__ temp)
{
  const int lane = threadIdx.x & 31;
  const int wave = threadIdx.x >> 5;
  int idx = blockIdx.x * WPB + wave;            // B*H*S = 65536 exactly
  int s = idx & (SS - 1);
  int h = (idx >> 9) & (HH - 1);
  int b = idx >> 12;

  const int coff = h * HD + lane * 4;
  float qf[MM][4], kf[MM][4], vf[MM][4];
#pragma unroll
  for (int j = 0; j < MM; ++j) {
    long base = ((long)((b * MM + j) * SS + s)) * DD + coff;
    const bf16_t* qp = Q + base;
    const bf16_t* kp = K + base;
    const bf16_t* vp = V + base;
#pragma unroll
    for (int c = 0; c < 4; ++c) {
      qf[j][c] = (float)qp[c];
      kf[j][c] = (float)kp[c];
      vf[j][c] = (float)vp[c];
    }
  }

  float sc[MM][MM];
#pragma unroll
  for (int i = 0; i < MM; ++i)
#pragma unroll
    for (int j = 0; j < MM; ++j) {
      float p = 0.0f;
#pragma unroll
      for (int c = 0; c < 4; ++c) p += qf[i][c] * kf[j][c];
      sc[i][j] = wave_sum(p);
    }

  float invscale = 1.0f / (sqrtf((float)HD) * fabsf(temp[0]));
  float lb0 = logf(sigmoidf_(betas[0]) + 1e-6f);
  float lb1 = logf(sigmoidf_(betas[1]) + 1e-6f);
  float bm[MM][MM];
#pragma unroll
  for (int i = 0; i < MM; ++i)
#pragma unroll
    for (int j = 0; j < MM; ++j) bm[i][j] = (i == j) ? 0.0f : -1e9f;
  bm[0][2] = lb0; bm[2][0] = 0.0f;   // pair (0,2): beta gate on [0,2] only
  bm[1][3] = lb1; bm[3][1] = 0.0f;   // pair (1,3): beta gate on [1,3] only

#pragma unroll
  for (int i = 0; i < MM; ++i) {
    float mx = -3.4e38f;
#pragma unroll
    for (int j = 0; j < MM; ++j) {
      sc[i][j] = sc[i][j] * invscale + bm[i][j];
      mx = fmaxf(mx, sc[i][j]);
    }
    float sum = 0.0f;
#pragma unroll
    for (int j = 0; j < MM; ++j) { sc[i][j] = expf(sc[i][j] - mx); sum += sc[i][j]; }
    float inv = 1.0f / sum;
#pragma unroll
    for (int j = 0; j < MM; ++j) sc[i][j] *= inv;
  }

#pragma unroll
  for (int i = 0; i < MM; ++i) {
    long base = ((long)((b * MM + i) * SS + s)) * DD + coff;
    bf16_t* op = AO + base;
#pragma unroll
    for (int c = 0; c < 4; ++c) {
      float o = 0.0f;
#pragma unroll
      for (int j = 0; j < MM; ++j) o += sc[i][j] * vf[j][c];
      op[c] = (bf16_t)o;
    }
  }
}

// ---------------------------------------------------------------------------
// Launch
// ---------------------------------------------------------------------------
extern "C" void kernel_launch(void* const* d_in, const int* in_sizes, int n_in,
                              void* d_out, int out_size, void* d_ws, size_t ws_size,
                              hipStream_t stream) {
  (void)in_sizes; (void)n_in; (void)out_size; (void)ws_size;

  const float* x[4]   = {(const float*)d_in[0], (const float*)d_in[3],
                         (const float*)d_in[6], (const float*)d_in[9]};
  const float* w[4]   = {(const float*)d_in[1], (const float*)d_in[4],
                         (const float*)d_in[7], (const float*)d_in[10]};
  const float* bin[4] = {(const float*)d_in[2], (const float*)d_in[5],
                         (const float*)d_in[8], (const float*)d_in[11]};
  const float* mod_emb = (const float*)d_in[12];
  const float* alphas  = (const float*)d_in[13];
  const float* betas   = (const float*)d_in[14];
  const float* gw1     = (const float*)d_in[15];
  const float* gb1     = (const float*)d_in[16];
  const float* gw2     = (const float*)d_in[17];
  const float* gb2     = (const float*)d_in[18];
  const float* wq = (const float*)d_in[19]; const float* bq = (const float*)d_in[20];
  const float* wk = (const float*)d_in[21]; const float* bk = (const float*)d_in[22];
  const float* wv = (const float*)d_in[23]; const float* bv = (const float*)d_in[24];
  const float* out_w = (const float*)d_in[25];
  const float* out_b = (const float*)d_in[26];
  const float* temp  = (const float*)d_in[27];
  float* out = (float*)d_out;

  // Workspace carve-out (~510 MB)
  char* ws = (char*)d_ws;
  size_t off = 0;
  auto carve = [&](size_t bytes) -> char* {
    char* p = ws + off;
    off += (bytes + 255) & ~(size_t)255;
    return p;
  };
  const size_t NROWS = (size_t)BB * MM * SS;                 // 32768
  const int dims[4] = {768, 1280, 2048, 2048};
  const int seqs[4] = {77, 77, 512, 512};

  float*  p32  = (float*) carve(NROWS * DD * sizeof(float)); // padded proj, f32
  float*  g1   = (float*) carve(NROWS * 256 * sizeof(float));// gate hidden, f32
  bf16_t* st   = (bf16_t*)carve(NROWS * DD * 2);             // stacked bf16
  bf16_t* qb   = (bf16_t*)carve(NROWS * DD * 2);
  bf16_t* kbuf = (bf16_t*)carve(NROWS * DD * 2);
  bf16_t* vb   = (bf16_t*)carve(NROWS * DD * 2);
  bf16_t* ao   = (bf16_t*)carve(NROWS * DD * 2);
  // packed bf16 weight copies (WMMA B-fragment tile order)
  bf16_t* wpk[4];
  for (int m = 0; m < 4; ++m) wpk[m] = (bf16_t*)carve((size_t)dims[m] * DD * 2);
  bf16_t* gw1pk = (bf16_t*)carve((size_t)MM * DD * 256 * 2);
  bf16_t* wqpk  = (bf16_t*)carve((size_t)DD * DD * 2);
  bf16_t* wkpk  = (bf16_t*)carve((size_t)DD * DD * 2);
  bf16_t* wvpk  = (bf16_t*)carve((size_t)DD * DD * 2);
  bf16_t* owpk  = (bf16_t*)carve((size_t)MM * DD * DD * 2);

  // 0) Weight prepass: convert + pack every GEMM weight once.
  auto pack = [&](const float* s, bf16_t* d, int K, int N) {
    int total = K * N;
    pack_w_kernel<<<(total + NTHREADS - 1) / NTHREADS, NTHREADS, 0, stream>>>(s, d, K, N);
  };
  for (int m = 0; m < 4; ++m) pack(w[m], wpk[m], dims[m], DD);
  for (int m = 0; m < 4; ++m)
    pack(gw1 + (size_t)m * DD * 256, gw1pk + (size_t)m * DD * 256, DD, 256);
  pack(wq, wqpk, DD, DD);
  pack(wk, wkpk, DD, DD);
  pack(wv, wvpk, DD, DD);
  for (int m = 0; m < 4; ++m)
    pack(out_w + (size_t)m * DD * DD, owpk + (size_t)m * DD * DD, DD, DD);

  // Padded rows of st must be exactly zero so K/V there equal bk/bv.
  (void)hipMemsetAsync(st, 0, NROWS * DD * 2, stream);

  auto gblocks = [](int R, int N) { return ((R + 255) / 256) * (N / 64); };

  // 1) Input projections: p = x_m @ w_m + b_m  -> p32 (mapped into [B,M,S,D])
  for (int m = 0; m < 4; ++m) {
    int R = BB * seqs[m];
    GemmMap am{seqs[m], m, 0};
    GemmMap cm{seqs[m], m, 1};
    gemm_wmma<float, 0><<<gblocks(R, DD), NTHREADS, 0, stream>>>(
        x[m], dims[m], am, wpk[m], bin[m], (void*)p32, DD, cm, R, DD, dims[m], 0);
  }

  // 2) Gate hidden: g1 = gelu(p @ gate_w1[m] + gate_b1[m])
  for (int m = 0; m < 4; ++m) {
    int R = BB * seqs[m];
    GemmMap am{seqs[m], m, 1};
    GemmMap cm{seqs[m], m, 1};
    gemm_wmma<float, 1><<<gblocks(R, 256), NTHREADS, 0, stream>>>(
        p32, DD, am, gw1pk + (size_t)m * DD * 256, gb1 + m * 256,
        (void*)g1, 256, cm, R, 256, DD, 0);
  }

  // 3) Gate scalar + modulation -> st (bf16)
  for (int m = 0; m < 4; ++m) {
    int R = BB * seqs[m];
    gate_mod_kernel<<<(R + WPB - 1) / WPB, NTHREADS, 0, stream>>>(
        p32, g1, gw2, gb2, alphas, mod_emb, st, seqs[m], m, R);
  }

  // 4) QKV projections over all 32768 rows (padded rows give bias vectors)
  {
    GemmMap dm{SS, 0, 0};
    gemm_wmma<bf16_t, 2><<<gblocks((int)NROWS, DD), NTHREADS, 0, stream>>>(
        st, DD, dm, wqpk, bq, (void*)qb, DD, dm, (int)NROWS, DD, DD, 0);
    gemm_wmma<bf16_t, 2><<<gblocks((int)NROWS, DD), NTHREADS, 0, stream>>>(
        st, DD, dm, wkpk, bk, (void*)kbuf, DD, dm, (int)NROWS, DD, DD, 0);
    gemm_wmma<bf16_t, 2><<<gblocks((int)NROWS, DD), NTHREADS, 0, stream>>>(
        st, DD, dm, wvpk, bv, (void*)vb, DD, dm, (int)NROWS, DD, DD, 0);
  }

  // 5) Per-position 4x4 cross-modality attention -> ao (bf16)
  attn_kernel<<<(BB * HH * SS) / WPB, NTHREADS, 0, stream>>>(qb, kbuf, vb, ao, betas, temp);

  // 6) Per-modality out projection + sequence mask -> d_out (f32)
  for (int m = 0; m < 4; ++m) {
    GemmMap am{SS, m, 1};
    GemmMap cm{SS, m, 1};
    gemm_wmma<bf16_t, 3><<<gblocks(BB * SS, DD), NTHREADS, 0, stream>>>(
        ao, DD, am, owpk + (size_t)m * DD * DD, out_b + m * DD,
        (void*)out, DD, cm, BB * SS, DD, DD, seqs[m]);
  }
}
